// RMoELayer_80161269612566
// MI455X (gfx1250) — matrix operand order repaired and checked by reading, not compiled, not run on hardware
//
#include <hip/hip_runtime.h>
#include <hip/hip_bf16.h>

typedef __bf16 bf16_t;
typedef bf16_t v16bf __attribute__((ext_vector_type(16)));
typedef float  v8f   __attribute__((ext_vector_type(8)));
typedef int    v4i   __attribute__((ext_vector_type(4)));

// Problem constants
constexpr int D_MODEL = 512;
constexpr int D_FF    = 2048;
constexpr int N_EXP   = 16;
constexpr int P_DIM   = 128;
constexpr int T_TOK   = 4096;   // B*S
constexpr int TOPK    = 2;

constexpr int SLOT_CAP  = T_TOK * TOPK + N_EXP * 16;   // 8448 (worst-case padded slots)
constexpr int TILE_CAP  = SLOT_CAP / 16;               // 528

// Workspace layout (bytes)
constexpr size_t OFF_META = 0;                                              // 64 ints
constexpr size_t OFF_XBF  = 256;
constexpr size_t OFF_W1T  = OFF_XBF + (size_t)T_TOK * D_MODEL * 2;
constexpr size_t OFF_W2T  = OFF_W1T + (size_t)N_EXP * D_MODEL * D_FF * 2;
constexpr size_t OFF_TOKE = OFF_W2T + (size_t)N_EXP * D_FF * D_MODEL * 2;
constexpr size_t OFF_TOKW = OFF_TOKE + (size_t)T_TOK * TOPK * 4;
constexpr size_t OFF_STOK = OFF_TOKW + (size_t)T_TOK * TOPK * 4;
constexpr size_t OFF_SW   = OFF_STOK + (size_t)SLOT_CAP * 4;
constexpr size_t OFF_HBF  = OFF_SW   + (size_t)SLOT_CAP * 4;
// total ~106 MB

// meta indices: [0..15] counts, [16..32] slotBase (17), [33..48] fill, [49] tilesTotal

// ---------------------------------------------------------------- CDNA5 async copy helpers
__device__ inline void async_copy_b128(const bf16_t* gsrc, bf16_t* lds_dst) {
#if __has_builtin(__builtin_amdgcn_global_load_async_to_lds_b128)
  typedef __attribute__((address_space(1))) v4i* gp_t;   // global (addrspace 1)
  typedef __attribute__((address_space(3))) v4i* lp_t;   // LDS    (addrspace 3)
  __builtin_amdgcn_global_load_async_to_lds_b128((gp_t)gsrc, (lp_t)lds_dst, 0, 0);
#else
  asm volatile("global_load_async_to_lds_b128 %0, %1, off"
               :: "v"((__attribute__((address_space(3))) bf16_t*)lds_dst),
                  "v"((const __attribute__((address_space(1))) bf16_t*)gsrc)
               : "memory");
#endif
}

__device__ inline void wait_async() {
#if __has_builtin(__builtin_amdgcn_s_wait_asynccnt)
  __builtin_amdgcn_s_wait_asynccnt(0);
#else
  asm volatile("s_wait_asynccnt 0x0" ::: "memory");
#endif
}

// ---------------------------------------------------------------- init
__global__ void init_kernel(float* __restrict__ out, int* __restrict__ slotTok,
                            int* __restrict__ meta) {
  int i = blockIdx.x * blockDim.x + threadIdx.x;
  if (i < T_TOK * D_MODEL) out[i] = 0.0f;
  if (i < SLOT_CAP) slotTok[i] = -1;
  if (i < 64) meta[i] = 0;
}

// ---------------------------------------------------------------- x: fp32 -> bf16
__global__ void convert_x_kernel(const float* __restrict__ x, bf16_t* __restrict__ xbf) {
  long i = (long)blockIdx.x * blockDim.x + threadIdx.x;
  if (i < (long)T_TOK * D_MODEL) xbf[i] = (bf16_t)x[i];
}

// ---------------------------------------------------------------- W[e][K][N] fp32 -> WT[e][N][K] bf16
__global__ __launch_bounds__(256) void transpose_kernel(const float* __restrict__ W,
                                                        bf16_t* __restrict__ WT,
                                                        int K, int N) {
  __shared__ float tile[32][33];
  const int e  = blockIdx.z;
  const int k0 = blockIdx.y * 32, n0 = blockIdx.x * 32;
  const float* src = W + (long)e * K * N;
  bf16_t*      dst = WT + (long)e * N * K;
  const int tx = threadIdx.x & 31, ty = threadIdx.x >> 5;   // 32 x 8
#pragma unroll
  for (int i = 0; i < 32; i += 8)
    tile[ty + i][tx] = src[(long)(k0 + ty + i) * N + n0 + tx];
  __syncthreads();
#pragma unroll
  for (int i = 0; i < 32; i += 8)
    dst[(long)(n0 + ty + i) * K + k0 + tx] = (bf16_t)tile[tx][ty + i];
}

// ---------------------------------------------------------------- gating (proj + GRU + router + top2)
__global__ __launch_bounds__(128) void gate_kernel(
    const float* __restrict__ x,   const float* __restrict__ hprev,
    const float* __restrict__ Wp,  const float* __restrict__ bp,
    const float* __restrict__ Wih, const float* __restrict__ Whh,
    const float* __restrict__ bih, const float* __restrict__ bhh,
    const float* __restrict__ Wr,  const float* __restrict__ br,
    float* __restrict__ hout, int* __restrict__ meta,
    int* __restrict__ tokE, float* __restrict__ tokW) {
  __shared__ float xs[D_MODEL];
  __shared__ float xps[P_DIM];
  __shared__ float hps[P_DIM];
  __shared__ float hns[P_DIM];
  __shared__ float lg[N_EXP];
  const int t = blockIdx.x, j = threadIdx.x;

  for (int i = j; i < D_MODEL; i += 128) xs[i] = x[(long)t * D_MODEL + i];
  hps[j] = hprev[(long)t * P_DIM + j];
  __syncthreads();

  {
    float acc = bp[j];
    const float* w = Wp + (long)j * D_MODEL;
    for (int i = 0; i < D_MODEL; ++i) acc += w[i] * xs[i];
    xps[j] = acc;
  }
  __syncthreads();

  // GRU gates (torch order r,z,n)
  float xr = bih[j], xz = bih[P_DIM + j], xn = bih[2 * P_DIM + j];
  float hr = bhh[j], hz = bhh[P_DIM + j], hn = bhh[2 * P_DIM + j];
  const float* wr_ = Wih + (long)j * P_DIM;
  const float* wz_ = Wih + (long)(P_DIM + j) * P_DIM;
  const float* wn_ = Wih + (long)(2 * P_DIM + j) * P_DIM;
  const float* vr_ = Whh + (long)j * P_DIM;
  const float* vz_ = Whh + (long)(P_DIM + j) * P_DIM;
  const float* vn_ = Whh + (long)(2 * P_DIM + j) * P_DIM;
  for (int i = 0; i < P_DIM; ++i) {
    const float xv = xps[i], hv = hps[i];
    xr += wr_[i] * xv; xz += wz_[i] * xv; xn += wn_[i] * xv;
    hr += vr_[i] * hv; hz += vz_[i] * hv; hn += vn_[i] * hv;
  }
  const float r = 1.0f / (1.0f + __expf(-(xr + hr)));
  const float z = 1.0f / (1.0f + __expf(-(xz + hz)));
  const float n = tanhf(xn + r * hn);
  const float h = (1.0f - z) * n + z * hps[j];
  hns[j] = h;
  hout[(long)t * P_DIM + j] = h;   // second tuple output
  __syncthreads();

  if (j < N_EXP) {
    float a = br[j];
    const float* w = Wr + (long)j * P_DIM;
    for (int i = 0; i < P_DIM; ++i) a += w[i] * hns[i];
    lg[j] = a;
  }
  __syncthreads();

  if (j == 0) {
    int i0 = 0; float m0 = lg[0];
    for (int e = 1; e < N_EXP; ++e) if (lg[e] > m0) { m0 = lg[e]; i0 = e; }
    int i1 = 0; float m1 = -3.0e38f;
    for (int e = 0; e < N_EXP; ++e) if (e != i0 && lg[e] > m1) { m1 = lg[e]; i1 = e; }
    const float w0 = 1.0f / (1.0f + __expf(m1 - m0));   // renormalized top-2 softmax
    tokE[t * 2 + 0] = i0; tokW[t * 2 + 0] = w0;
    tokE[t * 2 + 1] = i1; tokW[t * 2 + 1] = 1.0f - w0;
    atomicAdd(&meta[i0], 1);
    atomicAdd(&meta[i1], 1);
  }
}

// ---------------------------------------------------------------- scan (tiny)
__global__ void scan_kernel(int* __restrict__ meta) {
  if (blockIdx.x == 0 && threadIdx.x == 0) {
    int base = 0;
    for (int e = 0; e < N_EXP; ++e) {
      meta[16 + e] = base;
      base += ((meta[e] + 15) / 16) * 16;
    }
    meta[16 + N_EXP] = base;
    meta[49] = base / 16;
  }
}

// ---------------------------------------------------------------- place pairs into expert-sorted slots
__global__ void place_kernel(const int* __restrict__ tokE, const float* __restrict__ tokW,
                             int* __restrict__ meta,
                             int* __restrict__ slotTok, float* __restrict__ slotW) {
  int p = blockIdx.x * blockDim.x + threadIdx.x;
  if (p >= T_TOK * TOPK) return;
  const int e   = tokE[p];
  const int pos = atomicAdd(&meta[33 + e], 1);
  const int s   = meta[16 + e] + pos;
  slotTok[s] = p >> 1;
  slotW[s]   = tokW[p];
}

// ---------------------------------------------------------------- WMMA fragment helpers
// 16-bit A 16x32 layout: per lane, K = 16*half + 0..15 (contiguous, matches B).
__device__ inline v16bf load_a_frag(const bf16_t* __restrict__ As, int lane, int kk, int stride) {
  const int r = lane & 15, half = lane >> 4;
  const uint4* row128 = (const uint4*)(As + r * stride + kk);   // 16B-aligned by construction
  union { uint4 q; bf16_t h[8]; } L, H;
  L.q = row128[2 * half];       // K 0..7   (half0) / K 16..23 (half1)
  H.q = row128[2 * half + 1];   // K 8..15  (half0) / K 24..31 (half1)
  v16bf a;
#pragma unroll
  for (int i = 0; i < 8; ++i) { a[i] = L.h[i]; a[8 + i] = H.h[i]; }
  return a;
}

// B 32x16 from pre-transposed weights WT[n][k]: per lane, 16 contiguous bf16 (32B vector load).
__device__ inline v16bf load_b_frag(const bf16_t* __restrict__ colbase) {
  return *(const v16bf*)colbase;
}

// ---------------------------------------------------------------- grouped GEMM 1: H = relu(Xg @ W1[e] + b1[e])
__global__ __launch_bounds__(512) void expert_ffn1(
    const bf16_t* __restrict__ xbf, const bf16_t* __restrict__ w1t,
    const float* __restrict__ b1, const int* __restrict__ meta,
    const int* __restrict__ slotTok, bf16_t* __restrict__ hbf) {
  constexpr int ASTR = D_MODEL + 8;
  __shared__ alignas(16) bf16_t As[16 * ASTR];
  const int tt = blockIdx.x;
  if (tt >= meta[49]) return;
  const int row0 = tt * 16;
  int e = 0;
#pragma unroll
  for (int i = 1; i < N_EXP; ++i) if (row0 >= meta[16 + i]) e = i;

  const int tid = threadIdx.x;
  {   // gather A tile (16 x 512) into LDS via async copies; zero dead rows
    const int r = tid >> 5, c = tid & 31;        // 32 threads x 16 elems per row
    const int tok = slotTok[row0 + r];
    bf16_t* dst = As + r * ASTR + c * 16;
    if (tok >= 0) {
      const bf16_t* src = xbf + (long)tok * D_MODEL + c * 16;
      async_copy_b128(src, dst);
      async_copy_b128(src + 8, dst + 8);
    } else {
#pragma unroll
      for (int i = 0; i < 16; ++i) dst[i] = (bf16_t)0.0f;
    }
  }
  wait_async();
  __syncthreads();

  const int wave = tid >> 5, lane = tid & 31;
  const int n0 = wave * 128;                     // 16 waves cover N = 2048
  const int nlane = lane & 15, khalf = (lane >> 4) * 16;
  const bf16_t* Wt = w1t + (long)e * D_FF * D_MODEL;   // [n][k], k contiguous
  v8f acc[8] = {};
  for (int kk = 0; kk < D_MODEL; kk += 32) {
    const v16bf a = load_a_frag(As, lane, kk, ASTR);
#pragma unroll
    for (int nt = 0; nt < 8; ++nt) {
      const bf16_t* col = Wt + (long)(n0 + nt * 16 + nlane) * D_MODEL + kk + khalf;
      __builtin_prefetch(col + 32, 0, 3);
      const v16bf b = load_b_frag(col);
      acc[nt] = __builtin_amdgcn_wmma_f32_16x16x32_bf16(
          false, a, false, b, (short)0, acc[nt], false, false);
    }
  }
  // epilogue: bias + relu -> bf16 H
  const int mh = (lane >> 4) * 8;
#pragma unroll
  for (int nt = 0; nt < 8; ++nt) {
    const int N = n0 + nt * 16 + nlane;
    const float bias = b1[e * D_FF + N];
#pragma unroll
    for (int j = 0; j < 8; ++j) {
      float v = acc[nt][j] + bias;
      v = v > 0.0f ? v : 0.0f;
      hbf[(long)(row0 + j + mh) * D_FF + N] = (bf16_t)v;
    }
  }
}

// ---------------------------------------------------------------- grouped GEMM 2: out += w * (H @ W2[e] + b2[e])
__global__ __launch_bounds__(512) void expert_ffn2(
    const bf16_t* __restrict__ hbf, const bf16_t* __restrict__ w2t,
    const float* __restrict__ b2, const int* __restrict__ meta,
    const int* __restrict__ slotTok, const float* __restrict__ slotW,
    float* __restrict__ out) {
  constexpr int ASTR = D_FF + 8;
  __shared__ alignas(16) bf16_t As[16 * ASTR];   // 16 x 2048 bf16 (~66 KB)
  const int tt = blockIdx.x;
  if (tt >= meta[49]) return;
  const int row0 = tt * 16;
  int e = 0;
#pragma unroll
  for (int i = 1; i < N_EXP; ++i) if (row0 >= meta[16 + i]) e = i;

  const int tid = threadIdx.x;
  {   // stage H tile into LDS via async copies (rows are valid for all live tiles)
    const int r = tid >> 5, c = tid & 31;        // 32 threads x 64 elems per row
    const bf16_t* src = hbf + (long)(row0 + r) * D_FF + c * 64;
    bf16_t* dst = As + r * ASTR + c * 64;
#pragma unroll
    for (int i = 0; i < 8; ++i) async_copy_b128(src + i * 8, dst + i * 8);
  }
  wait_async();
  __syncthreads();

  const int wave = tid >> 5, lane = tid & 31;
  const int n0 = wave * 32;                      // 16 waves cover N = 512
  const int nlane = lane & 15, khalf = (lane >> 4) * 16;
  const bf16_t* Wt = w2t + (long)e * D_FF * D_MODEL;   // [n][k], k contiguous
  v8f acc[2] = {};
  for (int kk = 0; kk < D_FF; kk += 32) {
    const v16bf a = load_a_frag(As, lane, kk, ASTR);
#pragma unroll
    for (int nt = 0; nt < 2; ++nt) {
      const bf16_t* col = Wt + (long)(n0 + nt * 16 + nlane) * D_FF + kk + khalf;
      __builtin_prefetch(col + 32, 0, 3);
      const v16bf b = load_b_frag(col);
      acc[nt] = __builtin_amdgcn_wmma_f32_16x16x32_bf16(
          false, a, false, b, (short)0, acc[nt], false, false);
    }
  }
  const int mh = (lane >> 4) * 8;
#pragma unroll
  for (int nt = 0; nt < 2; ++nt) {
    const int N = n0 + nt * 16 + nlane;
    const float bias = b2[e * D_MODEL + N];
#pragma unroll
    for (int j = 0; j < 8; ++j) {
      const int s = row0 + j + mh;
      const int tok = slotTok[s];
      if (tok >= 0) {
        atomicAdd(&out[(long)tok * D_MODEL + N], slotW[s] * (acc[nt][j] + bias));
      }
    }
  }
}

// ---------------------------------------------------------------- launch
extern "C" void kernel_launch(void* const* d_in, const int* in_sizes, int n_in,
                              void* d_out, int out_size, void* d_ws, size_t ws_size,
                              hipStream_t stream) {
  const float* x    = (const float*)d_in[0];
  const float* hpr  = (const float*)d_in[1];
  const float* Wp   = (const float*)d_in[2];
  const float* bp   = (const float*)d_in[3];
  const float* Wih  = (const float*)d_in[4];
  const float* Whh  = (const float*)d_in[5];
  const float* bih  = (const float*)d_in[6];
  const float* bhh  = (const float*)d_in[7];
  const float* Wr   = (const float*)d_in[8];
  const float* br   = (const float*)d_in[9];
  const float* W1   = (const float*)d_in[10];
  const float* b1   = (const float*)d_in[11];
  const float* W2   = (const float*)d_in[12];
  const float* b2   = (const float*)d_in[13];

  char* ws = (char*)d_ws;
  int*    meta    = (int*)(ws + OFF_META);
  bf16_t* xbf     = (bf16_t*)(ws + OFF_XBF);
  bf16_t* w1t     = (bf16_t*)(ws + OFF_W1T);
  bf16_t* w2t     = (bf16_t*)(ws + OFF_W2T);
  int*    tokE    = (int*)(ws + OFF_TOKE);
  float*  tokW    = (float*)(ws + OFF_TOKW);
  int*    slotTok = (int*)(ws + OFF_STOK);
  float*  slotW   = (float*)(ws + OFF_SW);
  bf16_t* hbf     = (bf16_t*)(ws + OFF_HBF);

  float* out  = (float*)d_out;                       // [T, 512]
  float* hout = out + (size_t)T_TOK * D_MODEL;       // [T, 128]

  init_kernel<<<(T_TOK * D_MODEL + 255) / 256, 256, 0, stream>>>(out, slotTok, meta);

  convert_x_kernel<<<(T_TOK * D_MODEL + 255) / 256, 256, 0, stream>>>(x, xbf);

  // W1[e][512][2048] -> w1t[e][2048][512] ; W2[e][2048][512] -> w2t[e][512][2048]
  transpose_kernel<<<dim3(D_FF / 32, D_MODEL / 32, N_EXP), 256, 0, stream>>>(W1, w1t, D_MODEL, D_FF);
  transpose_kernel<<<dim3(D_MODEL / 32, D_FF / 32, N_EXP), 256, 0, stream>>>(W2, w2t, D_FF, D_MODEL);

  gate_kernel<<<T_TOK, 128, 0, stream>>>(x, hpr, Wp, bp, Wih, Whh, bih, bhh,
                                         Wr, br, hout, meta, tokE, tokW);

  scan_kernel<<<1, 1, 0, stream>>>(meta);

  place_kernel<<<(T_TOK * TOPK + 255) / 256, 256, 0, stream>>>(tokE, tokW, meta,
                                                               slotTok, slotW);

  expert_ffn1<<<TILE_CAP, 512, 0, stream>>>(xbf, w1t, b1, meta, slotTok, hbf);

  expert_ffn2<<<TILE_CAP, 512, 0, stream>>>(hbf, w2t, b2, meta, slotTok, slotW, out);
}
